// GPT2SelfAttention_25709674234208
// MI455X (gfx1250) — matrix operand order repaired and checked by reading, not compiled, not run on hardware
//
#include <hip/hip_runtime.h>
#include <hip/hip_bf16.h>

typedef _Float16 half_t;
typedef __attribute__((ext_vector_type(16))) _Float16 v16h;
typedef __attribute__((ext_vector_type(8)))  _Float16 v8h;
typedef __attribute__((ext_vector_type(4)))  _Float16 v4h;
typedef __attribute__((ext_vector_type(2)))  __fp16   v2fp;   // builtin return type
typedef __attribute__((ext_vector_type(8)))  float    v8f;

#define B_  2
#define S_  2048
#define D_  1024
#define H_  16
#define HD_ 64

union V16 { v16h v; v8h h[2]; };
union V4  { v4h v; v2fp p[2]; };

static __device__ inline v8f wmma16(v16h a, v16h b, v8f c) {
  return __builtin_amdgcn_wmma_f32_16x16x32_f16(false, a, false, b, (short)0, c,
                                                false, false);
}

static __device__ inline v16h ld2x8(const half_t* p0, const half_t* p1) {
  V16 u;
  u.h[0] = *(const v8h*)p0;
  u.h[1] = *(const v8h*)p1;
  return u.v;
}

// packed fp32x4 -> f16x4 (v_cvt_pk_f16_f32 x2)
static __device__ inline v4h cvt4(float4 t) {
  V4 u;
  u.p[0] = __builtin_amdgcn_cvt_pkrtz(t.x, t.y);
  u.p[1] = __builtin_amdgcn_cvt_pkrtz(t.z, t.w);
  return u.v;
}

// ---------------------------------------------------------------------------
// Kernel 1: QKV projection. C[4096,3072] = X[4096,1024] @ Wqkv[1024,3072] + b
// Register double-buffered global->LDS staging; b128 fragment gathers.
// Block: 256 threads (8 waves). Tile 128x128, K-step 32.
// ---------------------------------------------------------------------------
__global__ __launch_bounds__(256) void qkv_gemm_kernel(
    const float* __restrict__ x, const float* __restrict__ w,
    const float* __restrict__ bias, half_t* __restrict__ qb,
    half_t* __restrict__ kb, half_t* __restrict__ vb) {
  __shared__ __align__(16) half_t Ah[128][40];    // [row][k], pitch 80B
  __shared__ __align__(16) half_t WhT[128][40];   // [col][k], transposed

  const int tid = threadIdx.x;
  const int wv = tid >> 5;
  const int ln = tid & 31;
  const int M = ln & 15;
  const int hi = ln >> 4;
  const int bm = blockIdx.y;
  const int bn = blockIdx.x;

  // per-thread tile coordinates (fixed across K-steps)
  const int ar = tid >> 3, ac4 = tid & 7;        // A: row, float4-col
  const int wr = tid >> 5, wc4 = tid & 31;       // W: k-row, float4-col
  const float* xp = x + (size_t)(bm * 128 + ar) * D_ + ac4 * 4;
  const float* wp = w + (size_t)wr * (3 * D_) + bn * 128 + wc4 * 4;

  v8f acc[8];
#pragma unroll
  for (int i = 0; i < 8; i++) acc[i] = v8f{};

  float4 xa[4], wa[4];
#pragma unroll
  for (int i = 0; i < 4; i++) {
    xa[i] = *(const float4*)(xp + (size_t)(i * 32) * D_);
    wa[i] = *(const float4*)(wp + (size_t)(i * 8) * (3 * D_));
  }

  for (int ko = 0; ko < D_; ko += 32) {
    // stage registers -> LDS (converted to f16)
#pragma unroll
    for (int i = 0; i < 4; i++)
      *(v4h*)&Ah[ar + i * 32][ac4 * 4] = cvt4(xa[i]);
#pragma unroll
    for (int i = 0; i < 4; i++) {
      v4h hh = cvt4(wa[i]);
      WhT[wc4 * 4 + 0][wr + i * 8] = hh[0];
      WhT[wc4 * 4 + 1][wr + i * 8] = hh[1];
      WhT[wc4 * 4 + 2][wr + i * 8] = hh[2];
      WhT[wc4 * 4 + 3][wr + i * 8] = hh[3];
    }
    __syncthreads();

    // issue next tile's global loads early (overlap with WMMAs)
    const int kn = (ko + 32 < D_) ? ko + 32 : ko;  // clamped redundant reload
#pragma unroll
    for (int i = 0; i < 4; i++) {
      xa[i] = *(const float4*)(xp + kn + (size_t)(i * 32) * D_);
      wa[i] = *(const float4*)(wp + (size_t)kn * (3 * D_) +
                               (size_t)(i * 8) * (3 * D_));
    }
    // L2 prefetch two steps ahead
    const int kp = (ko + 64 < D_) ? ko + 64 : ko;
    __builtin_prefetch(xp + kp, 0, 1);
    __builtin_prefetch(wp + (size_t)kp * (3 * D_), 0, 1);

    const half_t* arow = &Ah[wv * 16 + M][0];
    v16h a = ld2x8(arow + hi * 16, arow + hi * 16 + 8);
#pragma unroll
    for (int nt = 0; nt < 8; nt++) {
      const half_t* brow = &WhT[nt * 16 + M][hi * 16];
      v16h bf = ld2x8(brow, brow + 8);
      acc[nt] = wmma16(a, bf, acc[nt]);
    }
    __syncthreads();
  }

  // epilogue: bias add, scatter to Q/K/V [B,H,S,HD] as f16
#pragma unroll
  for (int nt = 0; nt < 8; nt++) {
    int n_global = bn * 128 + nt * 16 + M;
    float bv = bias[n_global];
    int which = n_global >> 10;
    int d = n_global & (D_ - 1);
    int h = d >> 6;
    int hd = d & 63;
    half_t* dst = (which == 0) ? qb : (which == 1) ? kb : vb;
#pragma unroll
    for (int r = 0; r < 8; r++) {
      int m_global = bm * 128 + wv * 16 + r + (hi ? 8 : 0);
      int b_i = m_global >> 11;
      int s = m_global & (S_ - 1);
      dst[(((size_t)(b_i * H_ + h) * S_ + s) << 6) + hd] =
          (half_t)(acc[nt][r] + bv);
    }
  }
}

// ---------------------------------------------------------------------------
// Kernel 2: causal flash attention per (b,h). 128 thr (4 waves), q-tile 64.
// Register double-buffered K/V tile staging.
// ---------------------------------------------------------------------------
__global__ __launch_bounds__(128) void attn_kernel(
    const half_t* __restrict__ q, const half_t* __restrict__ k,
    const half_t* __restrict__ v, half_t* __restrict__ ctx) {
  __shared__ __align__(16) half_t Kt[32][72];       // [key][d]
  __shared__ __align__(16) half_t VtT[64][40];      // [d][key]
  __shared__ __align__(16) half_t Pst[4][16][40];   // per-wave P staging

  const int tid = threadIdx.x;
  const int wv = tid >> 5;
  const int ln = tid & 31;
  const int M = ln & 15;
  const int hi = ln >> 4;
  const int bh = blockIdx.y;
  const int qbase = blockIdx.x * 64;

  const half_t* Qh = q + (size_t)bh * S_ * HD_;
  const half_t* Kh = k + (size_t)bh * S_ * HD_;
  const half_t* Vh = v + (size_t)bh * S_ * HD_;

  // Q fragments: contiguous 16B chunks -> global_load_b128
  const half_t* qrowp = Qh + (size_t)(qbase + wv * 16 + M) * HD_;
  v16h a0 = ld2x8(qrowp + hi * 16, qrowp + hi * 16 + 8);
  v16h a1 = ld2x8(qrowp + 32 + hi * 16, qrowp + 40 + hi * 16);

  v8f o[4];
#pragma unroll
  for (int i = 0; i < 4; i++) o[i] = v8f{};
  float mrow[8], lrow[8];
#pragma unroll
  for (int r = 0; r < 8; r++) { mrow[r] = -1e30f; lrow[r] = 0.f; }

  // per-thread K/V staging coordinates: 2 chunks of 8 halves
  const int cr0 = tid >> 3, cc0 = (tid & 7) * 8;     // chunk 0: rows 0..15
  const int cr1 = 16 + cr0;                          // chunk 1: rows 16..31
  const int ktmax = (qbase >> 5) + 2;

  v8h kreg[2], vreg[2];
  kreg[0] = *(const v8h*)(Kh + (size_t)cr0 * HD_ + cc0);
  kreg[1] = *(const v8h*)(Kh + (size_t)cr1 * HD_ + cc0);
  vreg[0] = *(const v8h*)(Vh + (size_t)cr0 * HD_ + cc0);
  vreg[1] = *(const v8h*)(Vh + (size_t)cr1 * HD_ + cc0);

  for (int kt = 0; kt < ktmax; kt++) {
    const int kb = kt * 32;
    // stage registers -> LDS
    *(v8h*)&Kt[cr0][cc0] = kreg[0];
    *(v8h*)&Kt[cr1][cc0] = kreg[1];
#pragma unroll
    for (int e = 0; e < 8; e++) {
      VtT[cc0 + e][cr0] = vreg[0][e];
      VtT[cc0 + e][cr1] = vreg[1][e];
    }
    __syncthreads();

    // issue next K/V tile loads early (clamped on last iteration)
    const int kbn = (kt + 1 < ktmax) ? kb + 32 : kb;
    kreg[0] = *(const v8h*)(Kh + (size_t)(kbn + cr0) * HD_ + cc0);
    kreg[1] = *(const v8h*)(Kh + (size_t)(kbn + cr1) * HD_ + cc0);
    vreg[0] = *(const v8h*)(Vh + (size_t)(kbn + cr0) * HD_ + cc0);
    vreg[1] = *(const v8h*)(Vh + (size_t)(kbn + cr1) * HD_ + cc0);

    // scores: two 16x16 key sub-tiles, accumulate over d (2x32)
    v8f sc[2];
#pragma unroll
    for (int sub = 0; sub < 2; sub++) {
      const half_t* krow = &Kt[sub * 16 + M][0];
      v16h b0 = ld2x8(krow + hi * 16, krow + hi * 16 + 8);
      v16h b1 = ld2x8(krow + 32 + hi * 16, krow + 40 + hi * 16);
      v8f z = v8f{};
      z = wmma16(a0, b0, z);
      sc[sub] = wmma16(a1, b1, z);
    }

    // scale + causal mask + online softmax update
#pragma unroll
    for (int r = 0; r < 8; r++) {
      int row = qbase + wv * 16 + r + (hi ? 8 : 0);
      float s0 = sc[0][r] * 0.125f;
      float s1 = sc[1][r] * 0.125f;
      if (kb + M > row) s0 = -10000.f;
      if (kb + 16 + M > row) s1 = -10000.f;
      float mx = fmaxf(s0, s1);
      mx = fmaxf(mx, __shfl_xor(mx, 8, 16));
      mx = fmaxf(mx, __shfl_xor(mx, 4, 16));
      mx = fmaxf(mx, __shfl_xor(mx, 2, 16));
      mx = fmaxf(mx, __shfl_xor(mx, 1, 16));
      float mnew = fmaxf(mrow[r], mx);
      float corr = __expf(mrow[r] - mnew);
      float p0 = __expf(s0 - mnew);
      float p1 = __expf(s1 - mnew);
      lrow[r] = lrow[r] * corr + p0 + p1;
      mrow[r] = mnew;
      sc[0][r] = p0;
      sc[1][r] = p1;
#pragma unroll
      for (int nt = 0; nt < 4; nt++) o[nt][r] *= corr;
    }

    // C-layout -> A-layout via per-wave LDS staging
#pragma unroll
    for (int r = 0; r < 8; r++) {
      int prow = r + (hi ? 8 : 0);
      Pst[wv][prow][M]      = (half_t)sc[0][r];
      Pst[wv][prow][16 + M] = (half_t)sc[1][r];
    }
    const half_t* prow = &Pst[wv][M][0];
    v16h pa = ld2x8(prow + hi * 16, prow + hi * 16 + 8);

    // O += P(16x32) @ V(32x64)
#pragma unroll
    for (int nt = 0; nt < 4; nt++) {
      const half_t* vrow = &VtT[nt * 16 + M][hi * 16];
      v16h bv = ld2x8(vrow, vrow + 8);
      o[nt] = wmma16(pa, bv, o[nt]);
    }
    __syncthreads();
  }

  // finalize: reduce l, normalize, store merged-head context
  const int b_i = bh >> 4;
  const int h = bh & 15;
#pragma unroll
  for (int r = 0; r < 8; r++) {
    float lt = lrow[r];
    lt += __shfl_xor(lt, 8, 16);
    lt += __shfl_xor(lt, 4, 16);
    lt += __shfl_xor(lt, 2, 16);
    lt += __shfl_xor(lt, 1, 16);
    float inv = 1.0f / lt;
    int row = qbase + wv * 16 + r + (hi ? 8 : 0);
    size_t base = ((size_t)(b_i * S_ + row) * D_) + h * HD_;
#pragma unroll
    for (int nt = 0; nt < 4; nt++)
      ctx[base + nt * 16 + M] = (half_t)(o[nt][r] * inv);
  }
}

// ---------------------------------------------------------------------------
// Kernel 3: output projection. out[4096,1024] = ctx_f16 @ Wout[1024,1024] + b
// ---------------------------------------------------------------------------
__global__ __launch_bounds__(256) void out_gemm_kernel(
    const half_t* __restrict__ ctxh, const float* __restrict__ w,
    const float* __restrict__ bias, float* __restrict__ out) {
  __shared__ __align__(16) half_t Ah[128][40];
  __shared__ __align__(16) half_t WhT[128][40];

  const int tid = threadIdx.x;
  const int wv = tid >> 5;
  const int ln = tid & 31;
  const int M = ln & 15;
  const int hi = ln >> 4;
  const int bm = blockIdx.y;
  const int bn = blockIdx.x;

  const int ar = tid >> 2, ac8 = (tid & 3) * 8;  // A: 4 chunks/row
  const int wr = tid >> 5, wc4 = tid & 31;
  const half_t* ap = ctxh + (size_t)(bm * 128 + ar) * D_ + ac8;
  const float* wp = w + (size_t)wr * D_ + bn * 128 + wc4 * 4;

  v8f acc[8];
#pragma unroll
  for (int i = 0; i < 8; i++) acc[i] = v8f{};

  v8h aa[2];
  float4 wa[4];
#pragma unroll
  for (int i = 0; i < 2; i++) aa[i] = *(const v8h*)(ap + (size_t)(i * 64) * D_);
#pragma unroll
  for (int i = 0; i < 4; i++) wa[i] = *(const float4*)(wp + (size_t)(i * 8) * D_);

  for (int ko = 0; ko < D_; ko += 32) {
#pragma unroll
    for (int i = 0; i < 2; i++) *(v8h*)&Ah[ar + i * 64][ac8] = aa[i];
#pragma unroll
    for (int i = 0; i < 4; i++) {
      v4h hh = cvt4(wa[i]);
      WhT[wc4 * 4 + 0][wr + i * 8] = hh[0];
      WhT[wc4 * 4 + 1][wr + i * 8] = hh[1];
      WhT[wc4 * 4 + 2][wr + i * 8] = hh[2];
      WhT[wc4 * 4 + 3][wr + i * 8] = hh[3];
    }
    __syncthreads();

    const int kn = (ko + 32 < D_) ? ko + 32 : ko;
#pragma unroll
    for (int i = 0; i < 2; i++)
      aa[i] = *(const v8h*)(ap + kn + (size_t)(i * 64) * D_);
#pragma unroll
    for (int i = 0; i < 4; i++)
      wa[i] = *(const float4*)(wp + (size_t)(kn + i * 8) * D_);
    const int kp = (ko + 64 < D_) ? ko + 64 : ko;
    __builtin_prefetch(ap + kp, 0, 1);
    __builtin_prefetch(wp + (size_t)kp * D_, 0, 1);

    const half_t* arow = &Ah[wv * 16 + M][0];
    v16h a = ld2x8(arow + hi * 16, arow + hi * 16 + 8);
#pragma unroll
    for (int nt = 0; nt < 8; nt++) {
      const half_t* brow = &WhT[nt * 16 + M][hi * 16];
      v16h bf = ld2x8(brow, brow + 8);
      acc[nt] = wmma16(a, bf, acc[nt]);
    }
    __syncthreads();
  }

#pragma unroll
  for (int nt = 0; nt < 8; nt++) {
    int n_global = bn * 128 + nt * 16 + M;
    float bv = bias[n_global];
#pragma unroll
    for (int r = 0; r < 8; r++) {
      int m_global = bm * 128 + wv * 16 + r + (hi ? 8 : 0);
      out[(size_t)m_global * D_ + n_global] = acc[nt][r] + bv;
    }
  }
}

// ---------------------------------------------------------------------------
extern "C" void kernel_launch(void* const* d_in, const int* in_sizes, int n_in,
                              void* d_out, int out_size, void* d_ws,
                              size_t ws_size, hipStream_t stream) {
  const float* x     = (const float*)d_in[0];
  const float* w_qkv = (const float*)d_in[1];
  const float* b_qkv = (const float*)d_in[2];
  const float* w_out = (const float*)d_in[3];
  const float* b_out = (const float*)d_in[4];
  float* out = (float*)d_out;

  const size_t perbuf = (size_t)B_ * H_ * S_ * HD_;  // 4,194,304 halves
  half_t* qh  = (half_t*)d_ws;
  half_t* kh  = qh + perbuf;
  half_t* vh  = kh + perbuf;
  half_t* ctx = vh + perbuf;

  dim3 g1(3 * D_ / 128, (B_ * S_) / 128);  // 24 x 32
  qkv_gemm_kernel<<<g1, dim3(256), 0, stream>>>(x, w_qkv, b_qkv, qh, kh, vh);

  dim3 g2(S_ / 64, B_ * H_);  // 32 x 32
  attn_kernel<<<g2, dim3(128), 0, stream>>>(qh, kh, vh, ctx);

  dim3 g3(D_ / 128, (B_ * S_) / 128);  // 8 x 32
  out_gemm_kernel<<<g3, dim3(256), 0, stream>>>(ctx, w_out, b_out, out);

  (void)in_sizes; (void)n_in; (void)out_size; (void)ws_size;
}
